// AdvancedGNN_45973329937097
// MI455X (gfx1250) — matrix operand order repaired and checked by reading, not compiled, not run on hardware
//
#include <hip/hip_runtime.h>
#include <hip/hip_bf16.h>
#include <math.h>

#define NN   8192
#define HDIM 128
#define EPS  1e-5f
#define KSPLIT 4
#define BS_STRIDE 40   // bf16 elems per staged B row: 32 data + 8 pad (80B, 16B-aligned)

typedef __bf16 bf16_t;
typedef __attribute__((ext_vector_type(16))) __bf16 v16bf;
typedef __attribute__((ext_vector_type(8)))  __bf16 v8bf;
typedef __attribute__((ext_vector_type(4)))  __bf16 v4bf;
typedef __attribute__((ext_vector_type(8)))  float  v8f;
typedef __attribute__((ext_vector_type(4)))  float  v4f;

// ---------------- WMMA helpers (CDNA5 16x16x32 bf16, f32 accum) ----------------

__device__ __forceinline__ v8f wmma_bf16(v16bf a, v16bf b, v8f c) {
  return __builtin_amdgcn_wmma_f32_16x16x32_bf16(false, a, false, b, (short)0, c,
                                                 false, false);
}

// A fragment 16x32 bf16 from row-major A (pointer pre-offset to (row0,k0)).
__device__ __forceinline__ v16bf frag_a_bf16(const bf16_t* __restrict__ A, long lda, int lane) {
  const int m = lane & 15, h = lane >> 4;
  const bf16_t* p = A + (size_t)m * lda + h * 16;
  v8bf lo = *(const v8bf*)p;
  v8bf hi = *(const v8bf*)(p + 8);
  v16bf r;
#pragma unroll
  for (int i = 0; i < 8; ++i) { r[i] = lo[i]; r[i + 8] = hi[i]; }
  return r;
}

// B fragment 32x16 bf16 from BT[N,K] row-major (pointer pre-offset to (col0,k0)).
__device__ __forceinline__ v16bf frag_b_bf16(const bf16_t* __restrict__ BT, long ldb, int lane) {
  const int n = lane & 15, ko = (lane >> 4) * 16;
  const bf16_t* p = BT + (size_t)n * ldb + ko;
  v8bf lo = *(const v8bf*)p;
  v8bf hi = *(const v8bf*)(p + 8);
  v16bf r;
#pragma unroll
  for (int i = 0; i < 8; ++i) { r[i] = lo[i]; r[i + 8] = hi[i]; }
  return r;
}

// B fragment from an LDS-staged 128xBS_STRIDE tile (row n, 32 k-values).
__device__ __forceinline__ v16bf frag_b_lds(const bf16_t* base, int n0, int lane) {
  const int n = n0 + (lane & 15), ko = (lane >> 4) * 16;
  const bf16_t* p = base + n * BS_STRIDE + ko;
  v8bf lo = *(const v8bf*)p;
  v8bf hi = *(const v8bf*)(p + 8);
  v16bf r;
#pragma unroll
  for (int i = 0; i < 8; ++i) { r[i] = lo[i]; r[i + 8] = hi[i]; }
  return r;
}

// ---------------- adjacency fp32 -> bf16 (once; 128MB result is L2-resident) ------
__global__ void __launch_bounds__(256) cvt_adj_kernel(
    const float* __restrict__ src, bf16_t* __restrict__ dst) {
  const size_t i = ((size_t)blockIdx.x * 256 + threadIdx.x) * 4;
  v4f f = *(const v4f*)(src + i);
  v4bf b;
#pragma unroll
  for (int c = 0; c < 4; ++c) b[c] = (bf16_t)f[c];
  *(v4bf*)(dst + i) = b;
}

// ---------------- Main aggregation GEMM ------------------------------------------
// part[ks][8192,128] = adj_bf16[rows, krange] @ msgT^T
// Block: 128 threads / 4 waves, 64 rows, one K quarter (2048).  Grid (128, 4).
// B tile (32x128, 8KB) staged to LDS via async DMA, double buffered; A read from
// the L2-resident bf16 adjacency copy (fits the 192MB global L2).
__global__ void __launch_bounds__(128) agg_gemm_kernel(
    const bf16_t* __restrict__ adjb, const bf16_t* __restrict__ msgT,
    float* __restrict__ part) {
  __shared__ bf16_t Bs[2][128 * BS_STRIDE];
  const int lane = threadIdx.x & 31;
  const int wave = threadIdx.x >> 5;
  const int row0 = (blockIdx.x * 4 + wave) * 16;
  const int k0   = blockIdx.y * (NN / KSPLIT);

  // Stage one 32x128 B tile: thread t copies row t (4 x 16B) via async-to-LDS DMA.
  auto stage = [&](int k, int buf) {
    const int t = threadIdx.x;
    const bf16_t* src = msgT + (size_t)t * NN + k;
    uint32_t dst = (uint32_t)(size_t)&Bs[buf][t * BS_STRIDE];
#pragma unroll
    for (int c = 0; c < 4; ++c) {
      asm volatile("global_load_async_to_lds_b128 %0, %1, off"
                   :: "v"(dst + c * 16), "v"(src + c * 8) : "memory");
    }
  };

  v8f acc[8] = {};
  stage(k0, 0);
  const bf16_t* pA = adjb + (size_t)row0 * NN;
  const int nsteps = (NN / KSPLIT) / 32;  // 64
  for (int i = 0; i < nsteps; ++i) {
    const int k = k0 + i * 32;
    if (i + 1 < nsteps) {
      stage(k + 32, (i + 1) & 1);
      asm volatile("s_wait_asynccnt 0x4" ::: "memory");  // previous tile done
    } else {
      asm volatile("s_wait_asynccnt 0x0" ::: "memory");
    }
    __syncthreads();
    v16bf a = frag_a_bf16(pA + k, NN, lane);
    if (i + 16 < nsteps)  // stream-ahead hint on the A stream
      __builtin_prefetch((const void*)(pA + (size_t)(lane & 15) * NN + k + 1024), 0, 1);
    const bf16_t* bb = &Bs[i & 1][0];
#pragma unroll
    for (int j = 0; j < 8; ++j)
      acc[j] = wmma_bf16(a, frag_b_lds(bb, j * 16, lane), acc[j]);
    __syncthreads();  // all waves done reading this buffer before it is re-staged
  }
  const int n = lane & 15, rs = (lane >> 4) * 8;
  float* dst = part + (size_t)blockIdx.y * NN * HDIM;
#pragma unroll
  for (int j = 0; j < 8; ++j)
#pragma unroll
    for (int g = 0; g < 8; ++g)
      dst[(size_t)(row0 + rs + g) * HDIM + j * 16 + n] = acc[j][g];
}

// Sum the 4 K-split partials, emit bf16 aggregate for the GRU GEMMs.
__global__ void __launch_bounds__(256) agg_reduce_kernel(
    const float* __restrict__ P, bf16_t* __restrict__ abf) {
  const size_t i = (size_t)blockIdx.x * 256 + threadIdx.x;
  const size_t S = (size_t)NN * HDIM;
  abf[i] = (bf16_t)(P[i] + P[i + S] + P[i + 2 * S] + P[i + 3 * S]);
}

// ---------------- Generic bf16 WMMA GEMM: C[M,N] = A[M,K] @ BT[N,K]^T + bias ------
__global__ void __launch_bounds__(256) gemm_bf16_kernel(
    const bf16_t* __restrict__ A, int lda, const bf16_t* __restrict__ BT, int ldb,
    const float* __restrict__ bias, float* __restrict__ C, int ldc,
    int tiles_m, int tiles_n, int K) {
  const int w = blockIdx.x * 8 + (threadIdx.x >> 5);
  if (w >= tiles_m * tiles_n) return;  // wave-uniform: EXEC all-ones for WMMA
  const int tm = w / tiles_n, tn = w % tiles_n;
  const int lane = threadIdx.x & 31;
  v8f acc = {};
  const bf16_t* pA = A + (size_t)tm * 16 * lda;
  const bf16_t* pB = BT + (size_t)tn * 16 * ldb;
  for (int k = 0; k < K; k += 32) {
    v16bf a = frag_a_bf16(pA + k, lda, lane);
    v16bf b = frag_b_bf16(pB + k, ldb, lane);
    acc = wmma_bf16(a, b, acc);
  }
  const int n = lane & 15, rs = (lane >> 4) * 8;
  const float bv = bias[tn * 16 + n];
#pragma unroll
  for (int g = 0; g < 8; ++g)
    C[(size_t)(tm * 16 + rs + g) * ldc + tn * 16 + n] = acc[g] + bv;
}

// ---------------- Embedding: x = relu(LN(nodes[8192,75] @ W[75,128] + b)) ---------
__global__ void __launch_bounds__(128) emb_kernel(
    const float* __restrict__ nf, const float* __restrict__ W,
    const float* __restrict__ b, const float* __restrict__ g,
    const float* __restrict__ beta, float* __restrict__ x,
    bf16_t* __restrict__ xbf) {
  __shared__ float xs[80];
  __shared__ float red[HDIM];
  const int node = blockIdx.x, t = threadIdx.x;
  if (t < 75) xs[t] = nf[(size_t)node * 75 + t];
  __syncthreads();
  float s = b[t];
  for (int i = 0; i < 75; ++i) s = fmaf(xs[i], W[i * HDIM + t], s);
  red[t] = s; __syncthreads();
  for (int st = 64; st; st >>= 1) { if (t < st) red[t] += red[t + st]; __syncthreads(); }
  float mu = red[0] * (1.f / HDIM); __syncthreads();
  float d = s - mu;
  red[t] = d * d; __syncthreads();
  for (int st = 64; st; st >>= 1) { if (t < st) red[t] += red[t + st]; __syncthreads(); }
  float var = red[0] * (1.f / HDIM);
  float v = fmaxf(d * rsqrtf(var + EPS) * g[t] + beta[t], 0.f);
  x[(size_t)node * HDIM + t]   = v;
  xbf[(size_t)node * HDIM + t] = (bf16_t)v;
}

// ---------------- msg LN+relu, write transposed bf16 msgT[H, N] -------------------
__global__ void __launch_bounds__(128) msg_ln_kernel(
    const float* __restrict__ Y, const float* __restrict__ g,
    const float* __restrict__ beta, bf16_t* __restrict__ msgT) {
  __shared__ float red[HDIM];
  const int node = blockIdx.x, t = threadIdx.x;
  float s = Y[(size_t)node * HDIM + t];
  red[t] = s; __syncthreads();
  for (int st = 64; st; st >>= 1) { if (t < st) red[t] += red[t + st]; __syncthreads(); }
  float mu = red[0] * (1.f / HDIM); __syncthreads();
  float d = s - mu;
  red[t] = d * d; __syncthreads();
  for (int st = 64; st; st >>= 1) { if (t < st) red[t] += red[t + st]; __syncthreads(); }
  float var = red[0] * (1.f / HDIM);
  float v = fmaxf(d * rsqrtf(var + EPS) * g[t] + beta[t], 0.f);
  msgT[(size_t)t * NN + node] = (bf16_t)v;
}

// ---------------- attention: LN(64)+relu then score dot ---------------------------
__global__ void __launch_bounds__(64) att_ln_score_kernel(
    const float* __restrict__ Ya, const float* __restrict__ g,
    const float* __restrict__ beta, const float* __restrict__ W2,
    const float* __restrict__ b2, float* __restrict__ scores) {
  __shared__ float red[64];
  const int node = blockIdx.x, t = threadIdx.x;
  float s = Ya[(size_t)node * 64 + t];
  red[t] = s; __syncthreads();
  for (int st = 32; st; st >>= 1) { if (t < st) red[t] += red[t + st]; __syncthreads(); }
  float mu = red[0] * (1.f / 64.f); __syncthreads();
  float d = s - mu;
  red[t] = d * d; __syncthreads();
  for (int st = 32; st; st >>= 1) { if (t < st) red[t] += red[t + st]; __syncthreads(); }
  float var = red[0] * (1.f / 64.f); __syncthreads();
  float h = fmaxf(d * rsqrtf(var + EPS) * g[t] + beta[t], 0.f);
  red[t] = h * W2[t]; __syncthreads();
  for (int st = 32; st; st >>= 1) { if (t < st) red[t] += red[t + st]; __syncthreads(); }
  if (t == 0) scores[node] = red[0] + b2[0];
}

// ---------------- GRU pointwise combine -------------------------------------------
__global__ void __launch_bounds__(256) gru_combine_kernel(
    const float* __restrict__ Gi, const float* __restrict__ Gh,
    float* __restrict__ x, bf16_t* __restrict__ xbf) {
  const int idx = blockIdx.x * 256 + threadIdx.x;
  const int i = idx >> 7, j = idx & 127;
  const size_t base = (size_t)i * 384 + j;
  float r = 1.f / (1.f + expf(-(Gi[base] + Gh[base])));
  float z = 1.f / (1.f + expf(-(Gi[base + 128] + Gh[base + 128])));
  float nn = tanhf(Gi[base + 256] + r * Gh[base + 256]);
  float xn = (1.f - z) * nn + z * x[idx];
  x[idx]   = xn;
  xbf[idx] = (bf16_t)xn;
}

// ---------------- small converters ------------------------------------------------
__global__ void __launch_bounds__(256) cvt_bf16_kernel(
    const float* __restrict__ src, bf16_t* __restrict__ dst, int n) {
  int i = blockIdx.x * 256 + threadIdx.x;
  if (i < n) dst[i] = (bf16_t)src[i];
}
// src[fi,fo] -> dst[fo,fi] bf16
__global__ void __launch_bounds__(256) transpose_cvt_kernel(
    const float* __restrict__ src, bf16_t* __restrict__ dst, int fi, int fo) {
  int i = blockIdx.x * 256 + threadIdx.x;
  if (i >= fi * fo) return;
  int o = i / fi, r = i % fi;
  dst[i] = (bf16_t)src[(size_t)r * fo + o];
}

// ---------------- softmax over 8192 scores ----------------------------------------
__global__ void __launch_bounds__(1024) softmax_kernel(
    const float* __restrict__ scores, float* __restrict__ attn) {
  __shared__ float red[1024];
  const int t = threadIdx.x;
  float m = -1e30f;
  for (int i = t; i < NN; i += 1024) m = fmaxf(m, scores[i]);
  red[t] = m; __syncthreads();
  for (int st = 512; st; st >>= 1) { if (t < st) red[t] = fmaxf(red[t], red[t + st]); __syncthreads(); }
  m = red[0]; __syncthreads();
  float s = 0.f;
  for (int i = t; i < NN; i += 1024) s += expf(scores[i] - m);
  red[t] = s; __syncthreads();
  for (int st = 512; st; st >>= 1) { if (t < st) red[t] += red[t + st]; __syncthreads(); }
  float inv = 1.f / red[0];
  for (int i = t; i < NN; i += 1024) attn[i] = expf(scores[i] - m) * inv;
}

// ---------------- attention pooling: pooled[j] = sum_i attn[i]*x[i,j] -------------
__global__ void __launch_bounds__(256) pooled_kernel(
    const float* __restrict__ x, const float* __restrict__ attn,
    float* __restrict__ pooled) {
  __shared__ float red[256];
  const int j = blockIdx.x, t = threadIdx.x;
  float s = 0.f;
  for (int i = t; i < NN; i += 256) s = fmaf(attn[i], x[(size_t)i * HDIM + j], s);
  red[t] = s; __syncthreads();
  for (int st = 128; st; st >>= 1) { if (t < st) red[t] += red[t + st]; __syncthreads(); }
  if (t == 0) pooled[j] = red[0];
}

// ---------------- mol branch + output MLP head (single block) ---------------------
__global__ void __launch_bounds__(256) head_kernel(
    const float* __restrict__ mol, const float* __restrict__ pooled,
    const float* __restrict__ W1, const float* __restrict__ b1,
    const float* __restrict__ g1, const float* __restrict__ be1,
    const float* __restrict__ W2, const float* __restrict__ b2,
    const float* __restrict__ Wo1, const float* __restrict__ bo1,
    const float* __restrict__ go1, const float* __restrict__ beo1,
    const float* __restrict__ Wo2, const float* __restrict__ bo2,
    const float* __restrict__ go2, const float* __restrict__ beo2,
    const float* __restrict__ Wo3, const float* __restrict__ bo3,
    float* __restrict__ out) {
  __shared__ float red[256];
  __shared__ float fe[192];
  __shared__ float act[128];
  __shared__ float act2[64];
  const int t = threadIdx.x;
  float v = 0.f;
  if (t < 64) { v = b1[t]; for (int i = 0; i < 9; ++i) v = fmaf(mol[i], W1[i * 64 + t], v); }
  red[t] = (t < 64) ? v : 0.f; __syncthreads();
  for (int st = 128; st; st >>= 1) { if (t < st) red[t] += red[t + st]; __syncthreads(); }
  float mu = red[0] * (1.f / 64.f); __syncthreads();
  float d = v - mu;
  red[t] = (t < 64) ? d * d : 0.f; __syncthreads();
  for (int st = 128; st; st >>= 1) { if (t < st) red[t] += red[t + st]; __syncthreads(); }
  float var = red[0] * (1.f / 64.f); __syncthreads();
  if (t < 64) act[t] = fmaxf(d * rsqrtf(var + EPS) * g1[t] + be1[t], 0.f);
  __syncthreads();
  if (t < 64) {
    float m2 = b2[t];
    for (int i = 0; i < 64; ++i) m2 = fmaf(act[i], W2[i * 64 + t], m2);
    fe[128 + t] = m2;
  }
  if (t < 128) fe[t] = pooled[t];
  __syncthreads();
  v = 0.f;
  if (t < 128) { v = bo1[t]; for (int i = 0; i < 192; ++i) v = fmaf(fe[i], Wo1[i * 128 + t], v); }
  red[t] = (t < 128) ? v : 0.f; __syncthreads();
  for (int st = 128; st; st >>= 1) { if (t < st) red[t] += red[t + st]; __syncthreads(); }
  mu = red[0] * (1.f / 128.f); __syncthreads();
  d = v - mu;
  red[t] = (t < 128) ? d * d : 0.f; __syncthreads();
  for (int st = 128; st; st >>= 1) { if (t < st) red[t] += red[t + st]; __syncthreads(); }
  var = red[0] * (1.f / 128.f); __syncthreads();
  if (t < 128) act[t] = fmaxf(d * rsqrtf(var + EPS) * go1[t] + beo1[t], 0.f);
  __syncthreads();
  v = 0.f;
  if (t < 64) { v = bo2[t]; for (int i = 0; i < 128; ++i) v = fmaf(act[i], Wo2[i * 64 + t], v); }
  red[t] = (t < 64) ? v : 0.f; __syncthreads();
  for (int st = 128; st; st >>= 1) { if (t < st) red[t] += red[t + st]; __syncthreads(); }
  mu = red[0] * (1.f / 64.f); __syncthreads();
  d = v - mu;
  red[t] = (t < 64) ? d * d : 0.f; __syncthreads();
  for (int st = 128; st; st >>= 1) { if (t < st) red[t] += red[t + st]; __syncthreads(); }
  var = red[0] * (1.f / 64.f); __syncthreads();
  if (t < 64) act2[t] = fmaxf(d * rsqrtf(var + EPS) * go2[t] + beo2[t], 0.f);
  __syncthreads();
  red[t] = (t < 64) ? act2[t] * Wo3[t] : 0.f; __syncthreads();
  for (int st = 128; st; st >>= 1) { if (t < st) red[t] += red[t + st]; __syncthreads(); }
  if (t == 0) out[0] = red[0] + bo3[0];
}

// ---------------- host orchestration ----------------------------------------------
extern "C" void kernel_launch(void* const* d_in, const int* in_sizes, int n_in,
                              void* d_out, int out_size, void* d_ws, size_t ws_size,
                              hipStream_t stream) {
  (void)in_sizes; (void)n_in; (void)out_size; (void)ws_size;
  const float* nf   = (const float*)d_in[0];
  const float* adj  = (const float*)d_in[1];
  const float* mol  = (const float*)d_in[2];
  const float* attW1 = (const float*)d_in[3];
  const float* attb1 = (const float*)d_in[4];
  const float* attW2 = (const float*)d_in[5];
  const float* attb2 = (const float*)d_in[6];
  const float* att_beta = (const float*)d_in[7];
  const float* att_g    = (const float*)d_in[8];
  const float *Whh[4], *Wih[4], *bhh[4], *bih[4], *Wm[4], *bm[4], *m_beta[4], *m_g[4];
  for (int L = 0; L < 4; ++L) {
    int b = 9 + L * 8;
    Whh[L] = (const float*)d_in[b + 0];
    Wih[L] = (const float*)d_in[b + 1];
    bhh[L] = (const float*)d_in[b + 2];
    bih[L] = (const float*)d_in[b + 3];
    Wm[L]  = (const float*)d_in[b + 4];
    bm[L]  = (const float*)d_in[b + 5];
    m_beta[L] = (const float*)d_in[b + 6];
    m_g[L]    = (const float*)d_in[b + 7];
  }
  const float* embW    = (const float*)d_in[41];
  const float* embb    = (const float*)d_in[42];
  const float* emb_beta= (const float*)d_in[43];
  const float* emb_g   = (const float*)d_in[44];
  const float* molW1   = (const float*)d_in[45];
  const float* molb1   = (const float*)d_in[46];
  const float* molW2   = (const float*)d_in[47];
  const float* molb2   = (const float*)d_in[48];
  const float* mol_beta= (const float*)d_in[49];
  const float* mol_g   = (const float*)d_in[50];
  const float* oW1 = (const float*)d_in[51];
  const float* ob1 = (const float*)d_in[52];
  const float* oW2 = (const float*)d_in[53];
  const float* ob2 = (const float*)d_in[54];
  const float* oW3 = (const float*)d_in[55];
  const float* ob3 = (const float*)d_in[56];
  const float* o1_beta = (const float*)d_in[57];
  const float* o1_g    = (const float*)d_in[58];
  const float* o2_beta = (const float*)d_in[59];
  const float* o2_g    = (const float*)d_in[60];

  // Outputs: x[8192,128] f32, out[1], attn[8192]
  float* x    = (float*)d_out;
  float* outv = x + (size_t)NN * HDIM;
  float* attn = outv + 1;

  // Workspace carving (256B aligned)
  char* w = (char*)d_ws;
  size_t off = 0;
  auto carve = [&](size_t bytes) { void* p = w + off; off += (bytes + 255) & ~(size_t)255; return p; };
  bf16_t* adjb = (bf16_t*)carve((size_t)NN * NN * 2);            // 128MB: fits 192MB L2
  bf16_t* xbf  = (bf16_t*)carve((size_t)NN * HDIM * 2);
  bf16_t* abf  = (bf16_t*)carve((size_t)NN * HDIM * 2);
  bf16_t* msgT = (bf16_t*)carve((size_t)NN * HDIM * 2);
  float*  Y    = (float*) carve((size_t)NN * HDIM * 4);          // reused as Ya [8192,64]
  float*  P    = (float*) carve((size_t)KSPLIT * NN * HDIM * 4); // K-split partials
  float*  Gi   = (float*) carve((size_t)NN * 384 * 4);
  float*  Gh   = (float*) carve((size_t)NN * 384 * 4);
  float*  scores = (float*)carve((size_t)NN * 4);
  float*  pooled = (float*)carve(HDIM * 4);
  bf16_t* WmT_bf[4]; bf16_t* Wih_bf[4]; bf16_t* Whh_bf[4];
  for (int L = 0; L < 4; ++L) {
    WmT_bf[L] = (bf16_t*)carve(128 * 128 * 2);
    Wih_bf[L] = (bf16_t*)carve(384 * 128 * 2);
    Whh_bf[L] = (bf16_t*)carve(384 * 128 * 2);
  }
  bf16_t* WaT_bf = (bf16_t*)carve(64 * 128 * 2);

  // ---- adjacency conversion (once): after this, all 4 layers hit L2 ----
  cvt_adj_kernel<<<(int)(((size_t)NN * NN / 4) / 256), 256, 0, stream>>>(adj, adjb);

  // ---- weight prep ----
  for (int L = 0; L < 4; ++L) {
    transpose_cvt_kernel<<<(128 * 128 + 255) / 256, 256, 0, stream>>>(Wm[L], WmT_bf[L], 128, 128);
    cvt_bf16_kernel<<<(384 * 128 + 255) / 256, 256, 0, stream>>>(Wih[L], Wih_bf[L], 384 * 128);
    cvt_bf16_kernel<<<(384 * 128 + 255) / 256, 256, 0, stream>>>(Whh[L], Whh_bf[L], 384 * 128);
  }
  transpose_cvt_kernel<<<(128 * 64 + 255) / 256, 256, 0, stream>>>(attW1, WaT_bf, 128, 64);

  // ---- embedding ----
  emb_kernel<<<NN, 128, 0, stream>>>(nf, embW, embb, emb_g, emb_beta, x, xbf);

  // ---- conv layers ----
  for (int L = 0; L < 4; ++L) {
    gemm_bf16_kernel<<<512, 256, 0, stream>>>(xbf, 128, WmT_bf[L], 128, bm[L], Y, 128,
                                              NN / 16, 8, 128);
    msg_ln_kernel<<<NN, 128, 0, stream>>>(Y, m_g[L], m_beta[L], msgT);
    agg_gemm_kernel<<<dim3(NN / 64, KSPLIT), 128, 0, stream>>>(adjb, msgT, P);
    agg_reduce_kernel<<<NN * HDIM / 256, 256, 0, stream>>>(P, abf);
    gemm_bf16_kernel<<<1536, 256, 0, stream>>>(abf, 128, Wih_bf[L], 128, bih[L], Gi, 384,
                                               NN / 16, 24, 128);
    gemm_bf16_kernel<<<1536, 256, 0, stream>>>(xbf, 128, Whh_bf[L], 128, bhh[L], Gh, 384,
                                               NN / 16, 24, 128);
    gru_combine_kernel<<<NN * HDIM / 256, 256, 0, stream>>>(Gi, Gh, x, xbf);
  }

  // ---- attention pooling ----
  gemm_bf16_kernel<<<256, 256, 0, stream>>>(xbf, 128, WaT_bf, 128, attb1, Y, 64,
                                            NN / 16, 4, 128);
  att_ln_score_kernel<<<NN, 64, 0, stream>>>(Y, att_g, att_beta, attW2, attb2, scores);
  softmax_kernel<<<1, 1024, 0, stream>>>(scores, attn);
  pooled_kernel<<<HDIM, 256, 0, stream>>>(x, attn, pooled);

  // ---- mol branch + output MLP ----
  head_kernel<<<1, 256, 0, stream>>>(mol, pooled,
                                     molW1, molb1, mol_g, mol_beta, molW2, molb2,
                                     oW1, ob1, o1_g, o1_beta, oW2, ob2, o2_g, o2_beta,
                                     oW3, ob3, outv);
}
